// LR_PINN_phase1_48155173323486
// MI455X (gfx1250) — matrix operand order repaired and checked by reading, not compiled, not run on hardware
//
#include <hip/hip_runtime.h>

#define HDIM 64
#define BATCH 2097152
#define WAVES_PER_BLOCK 8
#define NMAT 11
#define FRAG_HALVES 4096      // one 64x64 matrix in B-fragment order (f16)
#define ACT_STRIDE 72         // halves per activation row (64 + 8 pad -> bank rotate)

typedef __attribute__((ext_vector_type(16))) _Float16 v16h;
typedef __attribute__((ext_vector_type(8)))  _Float16 v8h;
typedef __attribute__((ext_vector_type(8)))  float    v8f;

// tanh(x) = 1 - 2/(exp(2x)+1): v_exp_f32 + v_rcp_f32, co-executes with WMMA
__device__ __forceinline__ float tanh_fast(float v) {
    float e = __expf(2.0f * v);
    return 1.0f - 2.0f * __builtin_amdgcn_rcpf(e + 1.0f);
}

// B fragment: 16 contiguous halves per lane -> two ds_load_b128
__device__ __forceinline__ v16h lds_frag(const _Float16* p) {
    union { v16h v; v8h h[2]; } u;
    u.h[0] = *(const v8h*)(p);
    u.h[1] = *(const v8h*)(p + 8);
    return u.v;
}

// D = A(16x64) * W(64x64): 4 N-tiles x 2 K-chunks of v_wmma_f32_16x16x32_f16
__device__ __forceinline__ void mm64(const v16h A[2], const _Float16* wf, int lane,
                                     v8f acc[4]) {
#pragma unroll
    for (int t = 0; t < 4; ++t) {
        v8f c = {0.f, 0.f, 0.f, 0.f, 0.f, 0.f, 0.f, 0.f};
#pragma unroll
        for (int kc = 0; kc < 2; ++kc) {
            v16h b = lds_frag(wf + ((kc * 4 + t) * 32 + lane) * 16);
            c = __builtin_amdgcn_wmma_f32_16x16x32_f16(
                    false, A[kc], false, b, (short)0, c, false, false);
        }
        acc[t] = c;
    }
}

// ---- D-layout (f32 regs) -> per-wave LDS scratch (f16) with activation ----
__device__ __forceinline__ void store_tanh_bias(const v8f acc[4], const float bias[4],
                                                _Float16* myact, int lane) {
    const int hi = lane >> 4, nlo = lane & 15;
#pragma unroll
    for (int t = 0; t < 4; ++t)
#pragma unroll
        for (int r = 0; r < 8; ++r)
            myact[(r + hi * 8) * ACT_STRIDE + t * 16 + nlo] =
                (_Float16)tanh_fast(acc[t][r] + bias[t]);
}

__device__ __forceinline__ void store_tanh(const v8f acc[4], _Float16* myact, int lane) {
    const int hi = lane >> 4, nlo = lane & 15;
#pragma unroll
    for (int t = 0; t < 4; ++t)
#pragma unroll
        for (int r = 0; r < 8; ++r)
            myact[(r + hi * 8) * ACT_STRIDE + t * 16 + nlo] =
                (_Float16)tanh_fast(acc[t][r]);
}

__device__ __forceinline__ void store_mul(const v8f acc[4], const float al[4][8],
                                          _Float16* myact, int lane) {
    const int hi = lane >> 4, nlo = lane & 15;
#pragma unroll
    for (int t = 0; t < 4; ++t)
#pragma unroll
        for (int r = 0; r < 8; ++r)
            myact[(r + hi * 8) * ACT_STRIDE + t * 16 + nlo] =
                (_Float16)(acc[t][r] * al[t][r]);
}

// LDS scratch -> A fragments (ISA 16-bit A layout: K = e + 8*(e>=8) + 8*laneHi)
__device__ __forceinline__ void load_A(const _Float16* myact, int lane, v16h A[2]) {
    const int hi = lane >> 4, m = lane & 15;
#pragma unroll
    for (int c = 0; c < 2; ++c) {
        const _Float16* p = myact + m * ACT_STRIDE + c * 32 + hi * 8;
        union { v16h v; v8h h[2]; } u;
        u.h[0] = *(const v8h*)(p);
        u.h[1] = *(const v8h*)(p + 16);
        A[c] = u.v;
    }
}

extern "C" __global__ void __launch_bounds__(256, 1)
pinn_fused(const float* __restrict__ gx, const float* __restrict__ gt,
           const float* __restrict__ gbeta, const float* __restrict__ gnu,
           const float* __restrict__ grho,
           const float* __restrict__ start_w, const float* __restrict__ start_b,
           const float* __restrict__ end_w, const float* __restrict__ end_b,
           const float* __restrict__ m1_w, const float* __restrict__ m1_b,
           const float* __restrict__ m2_w, const float* __restrict__ m2_b,
           const float* __restrict__ m3_w, const float* __restrict__ m3_b,
           const float* __restrict__ a0_w, const float* __restrict__ a0_b,
           const float* __restrict__ a1_w, const float* __restrict__ a1_b,
           const float* __restrict__ a2_w, const float* __restrict__ a2_b,
           const float* __restrict__ col0, const float* __restrict__ col1,
           const float* __restrict__ col2, const float* __restrict__ row0,
           const float* __restrict__ row1, const float* __restrict__ row2,
           float* __restrict__ out)
{
    extern __shared__ __align__(16) _Float16 smem[];
    _Float16* wfrag = smem;                        // NMAT * FRAG_HALVES
    _Float16* act   = smem + NMAT * FRAG_HALVES;   // WAVES * 16 * ACT_STRIDE

    const int tid  = threadIdx.x;
    const int lane = tid & 31;
    const int wave = tid >> 5;
    const int hi   = lane >> 4;
    const int nlo  = lane & 15;

    // ---- stage all 11 weight matrices into LDS in B-fragment order ----
    {
        const float* Ws[NMAT] = { m2_w, m3_w, a0_w, a1_w, a2_w,
                                  col0, row0, col1, row1, col2, row2 };
        const int fl  = tid & 31;          // fragment lane
        const int ft  = (tid >> 5) & 3;    // N tile
        const int fc  = tid >> 7;          // K chunk
        const int fn  = ft * 16 + (fl & 15);
        const int fhi = fl >> 4;
#pragma unroll
        for (int mat = 0; mat < NMAT; ++mat) {
            const float* W = Ws[mat];
            _Float16* dst = wfrag + mat * FRAG_HALVES + tid * 16;
#pragma unroll
            for (int e = 0; e < 16; ++e) {
                const int k = fc * 32 + fhi * 8 + e + (e & 8);   // B layout: K = e + 8*(e>=8) + 8*laneHi
                dst[e] = (_Float16)W[k * HDIM + fn];
            }
        }
    }
    __syncthreads();

    _Float16* myact = act + wave * 16 * ACT_STRIDE;
    const long tileBase = ((long)blockIdx.x * WAVES_PER_BLOCK + wave) * 16;
    const long s = tileBase + nlo;

    const float xv = gx[s],    tv = gt[s];
    const float bv = gbeta[s], nv = gnu[s], rv = grho[s];

    // ---- front layers (2->64, 3->64) computed directly into A-fragment layout ----
    v16h hA[2], mA[2];
#pragma unroll
    for (int c = 0; c < 2; ++c) {
#pragma unroll
        for (int e = 0; e < 16; ++e) {
            const int f = c * 32 + e + (e & 8) + hi * 8;   // A-fragment feature index
            const float h = tanh_fast(xv * start_w[f] + tv * start_w[HDIM + f] + start_b[f]);
            hA[c][e] = (_Float16)h;
            const float m = tanh_fast(bv * m1_w[f] + nv * m1_w[HDIM + f] +
                                      rv * m1_w[2 * HDIM + f] + m1_b[f]);
            mA[c][e] = (_Float16)m;
        }
    }

    float biasM2[4], biasM3[4], biasA0[4], biasA1[4], biasA2[4];
#pragma unroll
    for (int t = 0; t < 4; ++t) {
        const int n = t * 16 + nlo;
        biasM2[t] = m2_b[n]; biasM3[t] = m3_b[n];
        biasA0[t] = a0_b[n]; biasA1[t] = a1_b[n]; biasA2[t] = a2_b[n];
    }

    v8f acc[4];

    // ---- hypernetwork trunk ----
    mm64(mA, wfrag + 0 * FRAG_HALVES, lane, acc);
    store_tanh_bias(acc, biasM2, myact, lane);
    load_A(myact, lane, mA);

    mm64(mA, wfrag + 1 * FRAG_HALVES, lane, acc);
    store_tanh_bias(acc, biasM3, myact, lane);
    load_A(myact, lane, mA);

    // ---- alphas kept in registers in D layout (matches h@col layout) ----
    float al0[4][8], al1[4][8], al2[4][8];
    mm64(mA, wfrag + 2 * FRAG_HALVES, lane, acc);
#pragma unroll
    for (int t = 0; t < 4; ++t)
#pragma unroll
        for (int r = 0; r < 8; ++r) al0[t][r] = fmaxf(acc[t][r] + biasA0[t], 0.f);
    mm64(mA, wfrag + 3 * FRAG_HALVES, lane, acc);
#pragma unroll
    for (int t = 0; t < 4; ++t)
#pragma unroll
        for (int r = 0; r < 8; ++r) al1[t][r] = fmaxf(acc[t][r] + biasA1[t], 0.f);
    mm64(mA, wfrag + 4 * FRAG_HALVES, lane, acc);
#pragma unroll
    for (int t = 0; t < 4; ++t)
#pragma unroll
        for (int r = 0; r < 8; ++r) al2[t][r] = fmaxf(acc[t][r] + biasA2[t], 0.f);

    // ---- main network: h = tanh(((h@col)*a)@row), 3 layers ----
    mm64(hA, wfrag + 5 * FRAG_HALVES, lane, acc);      // h @ col0
    store_mul(acc, al0, myact, lane);
    load_A(myact, lane, hA);
    mm64(hA, wfrag + 6 * FRAG_HALVES, lane, acc);      // @ row0
    store_tanh(acc, myact, lane);
    load_A(myact, lane, hA);

    mm64(hA, wfrag + 7 * FRAG_HALVES, lane, acc);      // h @ col1
    store_mul(acc, al1, myact, lane);
    load_A(myact, lane, hA);
    mm64(hA, wfrag + 8 * FRAG_HALVES, lane, acc);      // @ row1
    store_tanh(acc, myact, lane);
    load_A(myact, lane, hA);

    mm64(hA, wfrag + 9 * FRAG_HALVES, lane, acc);      // h @ col2
    store_mul(acc, al2, myact, lane);
    load_A(myact, lane, hA);
    mm64(hA, wfrag + 10 * FRAG_HALVES, lane, acc);     // @ row2 (final, stays in regs)

    // ---- head: out = tanh(h) @ end_w + end_b, cross-lane reduce over N ----
    float ew[4];
#pragma unroll
    for (int t = 0; t < 4; ++t) ew[t] = end_w[t * 16 + nlo];
    const float eb = end_b[0];
#pragma unroll
    for (int r = 0; r < 8; ++r) {
        float p = 0.f;
#pragma unroll
        for (int t = 0; t < 4; ++t)
            p += tanh_fast(acc[t][r]) * ew[t];
        p += __shfl_xor(p, 1, 32);
        p += __shfl_xor(p, 2, 32);
        p += __shfl_xor(p, 4, 32);
        p += __shfl_xor(p, 8, 32);
        if (nlo == 0) out[tileBase + hi * 8 + r] = p + eb;
    }
}

// outputs 1..6 of the tuple are the unmodified col/row matrices
extern "C" __global__ void __launch_bounds__(256)
copy_mats(const float* __restrict__ c0, const float* __restrict__ c1,
          const float* __restrict__ c2, const float* __restrict__ r0,
          const float* __restrict__ r1, const float* __restrict__ r2,
          float* __restrict__ dst)
{
    const int i = blockIdx.x * 256 + threadIdx.x;   // 0..24575
    const float* srcs[6] = { c0, c1, c2, r0, r1, r2 };
    dst[i] = srcs[i >> 12][i & 4095];
}

extern "C" void kernel_launch(void* const* d_in, const int* in_sizes, int n_in,
                              void* d_out, int out_size, void* d_ws, size_t ws_size,
                              hipStream_t stream) {
    (void)in_sizes; (void)n_in; (void)out_size; (void)d_ws; (void)ws_size;
    const float* x       = (const float*)d_in[0];
    const float* t       = (const float*)d_in[1];
    const float* beta    = (const float*)d_in[2];
    const float* nu      = (const float*)d_in[3];
    const float* rho     = (const float*)d_in[4];
    const float* start_w = (const float*)d_in[5];
    const float* start_b = (const float*)d_in[6];
    const float* end_w   = (const float*)d_in[7];
    const float* end_b   = (const float*)d_in[8];
    const float* m1_w    = (const float*)d_in[9];
    const float* m1_b    = (const float*)d_in[10];
    const float* m2_w    = (const float*)d_in[11];
    const float* m2_b    = (const float*)d_in[12];
    const float* m3_w    = (const float*)d_in[13];
    const float* m3_b    = (const float*)d_in[14];
    const float* a0_w    = (const float*)d_in[15];
    const float* a0_b    = (const float*)d_in[16];
    const float* a1_w    = (const float*)d_in[17];
    const float* a1_b    = (const float*)d_in[18];
    const float* a2_w    = (const float*)d_in[19];
    const float* a2_b    = (const float*)d_in[20];
    const float* col0    = (const float*)d_in[21];
    const float* col1    = (const float*)d_in[22];
    const float* col2    = (const float*)d_in[23];
    const float* row0    = (const float*)d_in[24];
    const float* row1    = (const float*)d_in[25];
    const float* row2    = (const float*)d_in[26];
    float* out = (float*)d_out;

    const int blocks = BATCH / (WAVES_PER_BLOCK * 16);   // 16384
    const size_t shmem =
        (size_t)(NMAT * FRAG_HALVES + WAVES_PER_BLOCK * 16 * ACT_STRIDE) * 2u;

    pinn_fused<<<blocks, 256, shmem, stream>>>(
        x, t, beta, nu, rho, start_w, start_b, end_w, end_b,
        m1_w, m1_b, m2_w, m2_b, m3_w, m3_b,
        a0_w, a0_b, a1_w, a1_b, a2_w, a2_b,
        col0, col1, col2, row0, row1, row2, out);

    copy_mats<<<(6 * 4096) / 256, 256, 0, stream>>>(
        col0, col1, col2, row0, row1, row2, out + (size_t)BATCH);
}
